// CoLAModel_32444182954835
// MI455X (gfx1250) — matrix operand order repaired and checked by reading, not compiled, not run on hardware
//
#include <hip/hip_runtime.h>
#include <hip/hip_bf16.h>

// ---------------------------------------------------------------------------
// Problem constants (from reference)
// ---------------------------------------------------------------------------
#define NB   16384          // B: number of graph blocks / anchors
#define NS   4              // S
#define NN   (NB*NS)        // N = 65536 nodes
#define ND   300            // D feature dim
#define NDP  320            // D padded to a multiple of 32 for WMMA K-chunks
#define NO   64             // O embedding dim
#define INV_TAU 2.0f        // 1/TAU, TAU=0.5
#define EPS_F 1e-5f

typedef __attribute__((ext_vector_type(16))) __bf16 v16bf;
typedef __attribute__((ext_vector_type(8)))  float  v8f;
typedef __attribute__((ext_vector_type(4)))  unsigned int u32x4;
typedef __attribute__((ext_vector_type(8)))  int i32x8;
typedef __attribute__((ext_vector_type(4)))  int i32x4;

#if defined(__has_builtin)
#if __has_builtin(__builtin_amdgcn_tensor_load_to_lds)
#define HAVE_TDM 1
#endif
#endif
#ifndef HAVE_TDM
#define HAVE_TDM 0
#endif

// ---------------------------------------------------------------------------
// WMMA helpers (gfx1250 wave32, V_WMMA_F32_16X16X32_BF16)
// ---------------------------------------------------------------------------
__device__ __forceinline__ v8f wmma_bf16(v16bf a, v16bf b, v8f c) {
  return __builtin_amdgcn_wmma_f32_16x16x32_bf16(
      false, a, false, b, (short)0, c, false, false);
}

union BfPack { uint4 u[2]; v16bf v; };

// A-tile 16x32 from row-major bf16 [row stride ldk, multiple of 32]: two
// contiguous 16-byte loads per lane (k0+8hi..+7 and k0+16+8hi..+7).
// Layout (ISA 7.12.2): lane m=l&15, hi=l>>4; VGPR j -> K=(j<4?2j:2j+8)+8*hi.
// Works for both global and LDS pointers (inlined; addrspace inferred).
__device__ __forceinline__ v16bf load_a_bf(const __bf16* base, int ldk,
                                           int row0, int k0) {
  const int lane = threadIdx.x & 31;
  const int m = lane & 15, hi = lane >> 4;
  const __bf16* rp = base + (size_t)(row0 + m) * (size_t)ldk + k0 + 8 * hi;
  BfPack t;
  t.u[0] = *(const uint4*)rp;
  t.u[1] = *(const uint4*)(rp + 16);
  return t.v;
}

// B-tile 32x16 bf16 where the logical KxN operand is stored N-major (N x ldk
// bf16, ldk multiple of 32): lane n reads K = k0+8hi..+7 and k0+16+8hi..+7.
__device__ __forceinline__ v16bf load_b_bf(const __bf16* base, int ldk,
                                           int col0, int k0) {
  const int lane = threadIdx.x & 31;
  const int n = lane & 15, hi = lane >> 4;
  const __bf16* cp = base + (size_t)(col0 + n) * (size_t)ldk + k0 + 8 * hi;
  BfPack t;
  t.u[0] = *(const uint4*)cp;
  t.u[1] = *(const uint4*)(cp + 16);
  return t.v;
}

// Store C tile (16x16 fp32) to row-major dst: row = r+8*hi, col = lane&15.
__device__ __forceinline__ void store_c_rm(float* __restrict__ dst, int ld,
                                           int row0, int col0, v8f c) {
  const int lane = threadIdx.x & 31;
  const int n = lane & 15, hi = lane >> 4;
#pragma unroll
  for (int r = 0; r < 8; ++r)
    dst[(size_t)(row0 + r + 8 * hi) * (size_t)ld + col0 + n] = c[r];
}

// ---------------------------------------------------------------------------
// K0: one-time bf16 repack of the reused weights.
//   WtB  : W  (300x64) -> transposed, K-padded (64 rows x 320 K) bf16
//   WsTB : Ws (64x64)  -> transposed (64 rows x 64 K) bf16
// ---------------------------------------------------------------------------
__global__ void k_cvt_w(const float* __restrict__ W, const float* __restrict__ Ws,
                        __bf16* __restrict__ WtB, __bf16* __restrict__ WsTB) {
  const int i = blockIdx.x * blockDim.x + threadIdx.x;
  if (i < NO * NDP) {
    const int n = i / NDP, k = i % NDP;
    WtB[i] = (k < ND) ? (__bf16)W[(size_t)k * NO + n] : (__bf16)0.0f;
  }
  const int j = i - NO * NDP;
  if (j >= 0 && j < NO * NO) {
    const int n = j / NO, k = j % NO;
    WsTB[j] = (__bf16)Ws[(size_t)k * NO + n];
  }
}

// ---------------------------------------------------------------------------
// K1: X = feat @ W.
//  - 40KB bf16 W panel staged into LDS via the Tensor Data Mover (wave 0
//    issues the D#, waits TENSORcnt; cooperative-copy fallback otherwise).
//  - The 16-row fp32 feature panel (single-use HBM stream) is staged ONCE per
//    block: coalesced fp32 loads -> bf16 -> zero-padded 16x320 LDS panel, so
//    the per-wave inner loop is pure ds_load_b128 + v_wmma (no cvt VALU).
// grid.x = NN/16, block = 128 (4 waves, one 16x16 col tile each)
// ---------------------------------------------------------------------------
__global__ void k_gemm_feat(const float* __restrict__ feat,
                            const __bf16* __restrict__ WtB,
                            float* __restrict__ X) {
  __shared__ __bf16 ldsW[NO * NDP];          // 40960 bytes
  __shared__ __bf16 ldsA[16 * NDP];          // 10240 bytes

  const int row0 = blockIdx.x * 16;

#if HAVE_TDM
  if (threadIdx.x < 32) {                    // wave 0 only issues the TDM op
    const unsigned long long ga = (unsigned long long)(size_t)WtB;
    const unsigned int lds_addr = (unsigned int)(size_t)(void*)ldsW;
    u32x4 g0;
    g0[0] = 1u;                                            // count=1, user mode
    g0[1] = lds_addr;                                      // lds_addr
    g0[2] = (unsigned int)(ga & 0xFFFFFFFFu);              // global_addr[31:0]
    g0[3] = (unsigned int)((ga >> 32) & 0x1FFFFFFu)        // global_addr[56:32]
            | (2u << 30);                                  // type=2 (image)
    i32x8 g1;
    g1[0] = 1 << 16;                 // workgroup_mask=0, data_size=1 (2 bytes)
    g1[1] = (int)(NDP << 16);        // tensor_dim0[15:0] = 320
    g1[2] = (int)(NO << 16);         // tensor_dim0[31:16]=0 | tensor_dim1 = 64
    g1[3] = (int)(NDP << 16);        // tensor_dim1[31:16]=0 | tile_dim0 = 320
    g1[4] = NO;                      // tile_dim1 = 64, tile_dim2 = 0
    g1[5] = NDP;                     // tensor_dim0_stride[31:0] = 320
    g1[6] = 0;                       // stride hi | tensor_dim1_stride lo
    g1[7] = 0;
    i32x4 z4 = {0, 0, 0, 0};
#if __clang_major__ >= 23
    i32x8 z8 = {0, 0, 0, 0, 0, 0, 0, 0};
    __builtin_amdgcn_tensor_load_to_lds(g0, g1, z4, z4, z8, 0);
#else
    __builtin_amdgcn_tensor_load_to_lds(g0, g1, z4, z4, 0);
#endif
    __builtin_amdgcn_s_wait_tensorcnt(0);
  }
#else
  for (int i = threadIdx.x; i < (NO * NDP) / 8; i += blockDim.x)
    ((uint4*)ldsW)[i] = ((const uint4*)WtB)[i];
#endif

  // Cooperative one-shot stage of the A panel: coalesced fp32 reads, convert
  // once, zero-pad K to 320.
  for (int i = threadIdx.x; i < 16 * NDP; i += blockDim.x) {
    const int r = i / NDP, k = i - r * NDP;
    const float v = (k < ND) ? feat[(size_t)(row0 + r) * ND + k] : 0.0f;
    ldsA[i] = (__bf16)v;
  }
  __syncthreads();

  const int col0 = (threadIdx.x >> 5) * 16;
  v8f c = {};
#pragma unroll
  for (int kc = 0; kc < NDP / 32; ++kc) {    // 10 chunks, tail is zero-padded
    v16bf a = load_a_bf(ldsA, NDP, 0, kc * 32);     // ds_load_b128 x2
    v16bf b = load_b_bf(ldsW, NDP, col0, kc * 32);  // ds_load_b128 x2
    c = wmma_bf16(a, b, c);
  }
  store_c_rm(X, NO, row0, col0, c);
}

// ---------------------------------------------------------------------------
// K2: per graph-block 4x4 weighted combine (anchor row zeroed => skip i=0),
// PReLU, mean-pool (emitted as bf16 GEMM operand); pos branch also emits
// l2norm(prelu(X[4b]+bias)) in fp32 (for final dot) and bf16 (score GEMM A).
// grid.x = NB/4, block = 256 (4 blocks x 64 features)
// ---------------------------------------------------------------------------
__global__ void k_block_reduce(const float* __restrict__ X,
                               const float* __restrict__ ew,
                               const float* __restrict__ bias,
                               const float* __restrict__ prelu_a,
                               __bf16* __restrict__ poolB,
                               float* __restrict__ anchF,
                               __bf16* __restrict__ anchB) {
  __shared__ float red[256];
  const int t = threadIdx.x;
  const int g = t >> 6;
  const int f = t & 63;
  const size_t b = (size_t)blockIdx.x * 4 + g;
  const float a = prelu_a[0];
  const float bf = bias[f];

  const float* xr = X + b * 4 * NO;
  const float x0 = xr[f], x1 = xr[NO + f], x2 = xr[2 * NO + f], x3 = xr[3 * NO + f];
  const float* w = ew + b * 16;

  float pv = 0.0f;
#pragma unroll
  for (int j = 0; j < 4; ++j) {
    float hj = w[4 + j] * x1 + w[8 + j] * x2 + w[12 + j] * x3 + bf;
    hj = (hj >= 0.0f) ? hj : a * hj;         // prelu
    pv += hj;
  }
  poolB[b * NO + f] = (__bf16)(pv * 0.25f);

  if (anchF) {                               // pos branch only (uniform)
    float av = x0 + bf;
    av = (av >= 0.0f) ? av : a * av;
    red[t] = av * av;
    __syncthreads();
#pragma unroll
    for (int s = 32; s > 0; s >>= 1) {
      if (f < s) red[t] += red[t + s];
      __syncthreads();
    }
    const float nrm = fmaxf(sqrtf(red[g * 64]), 1e-12f);
    const float v = av / nrm;
    anchF[b * NO + f] = v;
    anchB[b * NO + f] = (__bf16)v;
  }
}

// ---------------------------------------------------------------------------
// K3: emb = l2norm(pool @ Ws + bs). One wave per 16 rows; pure-bf16 operands,
// 8 WMMAs; bias + row l2norm via LDS; fp32 + bf16 outputs.
// grid.x = NB/16, block = 32
// ---------------------------------------------------------------------------
__global__ void k_pool_emb(const __bf16* __restrict__ poolB,
                           const __bf16* __restrict__ WsTB,
                           const float* __restrict__ bs,
                           float* __restrict__ embF,
                           __bf16* __restrict__ embB) {
  __shared__ float tile[16 * NO];
  __shared__ float norms[16];
  const int row0 = blockIdx.x * 16;
  const int lane = threadIdx.x & 31;
  const int n = lane & 15, hi = lane >> 4;

#pragma unroll
  for (int ct = 0; ct < 4; ++ct) {
    const int col0 = ct * 16;
    v8f c = {};
#pragma unroll
    for (int kc = 0; kc < 2; ++kc) {
      v16bf a = load_a_bf(poolB, NO, row0, kc * 32);
      v16bf b = load_b_bf(WsTB, NO, col0, kc * 32);
      c = wmma_bf16(a, b, c);
    }
#pragma unroll
    for (int r = 0; r < 8; ++r)
      tile[(r + 8 * hi) * NO + col0 + n] = c[r] + bs[col0 + n];
  }
  __syncthreads();
  if (lane < 16) {
    float s = 0.0f;
#pragma unroll
    for (int cc = 0; cc < NO; ++cc) { float v = tile[lane * NO + cc]; s += v * v; }
    norms[lane] = fmaxf(sqrtf(s), 1e-12f);
  }
  __syncthreads();
  for (int idx = lane; idx < 16 * NO; idx += 32) {
    const int r = idx >> 6, cc = idx & 63;
    const float v = tile[idx] / norms[r];
    embF[(size_t)(row0 + r) * NO + cc] = v;
    embB[(size_t)(row0 + r) * NO + cc] = (__bf16)v;
  }
}

// ---------------------------------------------------------------------------
// K4: neg_all[k] = sum_j exp( dot(anchorEmb[k], negEmb[j]) / TAU )
// Dominant compute: bf16 WMMA sweep over B=16384 cols; A tile in registers;
// hot loop = 4x global_load_b128 + 2x wmma + 8x v_exp_f32. Both operand
// tables are 2MB bf16 => L2-resident on the 192MB L2.
// grid.x = NB/16, block = 256
// ---------------------------------------------------------------------------
__global__ void k_score(const __bf16* __restrict__ anchB,
                        const __bf16* __restrict__ negB,
                        float* __restrict__ negAll) {
  __shared__ float rowsum[16];
  const int t = threadIdx.x;
  if (t < 16) rowsum[t] = 0.0f;
  __syncthreads();

  const int wave = t >> 5;
  const int lane = t & 31;
  const int n = lane & 15, hi = lane >> 4;
  const int row0 = blockIdx.x * 16;

  const v16bf a0 = load_a_bf(anchB, NO, row0, 0);
  const v16bf a1 = load_a_bf(anchB, NO, row0, 32);

  float acc[8];
#pragma unroll
  for (int r = 0; r < 8; ++r) acc[r] = 0.0f;

  for (int ct = wave; ct < NB / 16; ct += 8) {
    const int col0 = ct * 16;
    if (ct + 8 < NB / 16)
      __builtin_prefetch(negB + (size_t)(col0 + 128) * NO, 0, 1);
    v16bf b0 = load_b_bf(negB, NO, col0, 0);
    v16bf b1 = load_b_bf(negB, NO, col0, 32);
    v8f c = {};
    c = wmma_bf16(a0, b0, c);
    c = wmma_bf16(a1, b1, c);
#pragma unroll
    for (int r = 0; r < 8; ++r)
      acc[r] += __expf(c[r] * INV_TAU);
  }

#pragma unroll
  for (int m = 8; m > 0; m >>= 1) {
#pragma unroll
    for (int r = 0; r < 8; ++r)
      acc[r] += __shfl_xor(acc[r], m, 32);
  }
  if (n == 0) {
#pragma unroll
    for (int r = 0; r < 8; ++r)
      atomicAdd(&rowsum[r + 8 * hi], acc[r]);     // ds_add_f32
  }
  __syncthreads();
  if (t < 16) negAll[row0 + t] = rowsum[t];
}

// ---------------------------------------------------------------------------
// K5: final loss / pos_score / neg_score  (BETA=0 kills the generator path)
// ---------------------------------------------------------------------------
__global__ void k_final(const float* __restrict__ poolEmb,
                        const float* __restrict__ anchorEmb,
                        const float* __restrict__ negAll,
                        float* __restrict__ out) {
  const int k = blockIdx.x * blockDim.x + threadIdx.x;
  if (k >= NB) return;
  float d = 0.0f;
  const float* p = poolEmb + (size_t)k * NO;
  const float* q = anchorEmb + (size_t)k * NO;
#pragma unroll 8
  for (int f = 0; f < NO; ++f) d += p[f] * q[f];
  const float pos = __expf(d * INV_TAU);
  const float na = negAll[k];
  out[k]          = -__logf(pos / (na + EPS_F));   // loss_pool
  out[NB + k]     = pos;                           // pos_score
  out[2 * NB + k] = na / (float)NB;                // neg_score
}

// ---------------------------------------------------------------------------
// Host launcher
// ---------------------------------------------------------------------------
extern "C" void kernel_launch(void* const* d_in, const int* in_sizes, int n_in,
                              void* d_out, int out_size, void* d_ws, size_t ws_size,
                              hipStream_t stream) {
  const float* pos_feat = (const float*)d_in[0];
  const float* pos_w    = (const float*)d_in[3];
  const float* neg_feat = (const float*)d_in[4];
  const float* neg_w    = (const float*)d_in[7];
  const float* Wm       = (const float*)d_in[12];
  const float* bias     = (const float*)d_in[13];
  const float* prelu_a  = (const float*)d_in[14];
  const float* Ws       = (const float*)d_in[15];
  const float* bs       = (const float*)d_in[16];
  float* out = (float*)d_out;

  // workspace layout (bytes, 16B-aligned slices)
  char* p = (char*)d_ws;
  float*  Xp      = (float*)p;  p += (size_t)NN * NO * 4;       // 16 MB
  float*  Xn      = (float*)p;  p += (size_t)NN * NO * 4;       // 16 MB
  float*  pAnchF  = (float*)p;  p += (size_t)NB * NO * 4;       // 4 MB
  float*  pPoolE  = (float*)p;  p += (size_t)NB * NO * 4;       // 4 MB
  float*  nPoolE  = (float*)p;  p += (size_t)NB * NO * 4;       // 4 MB
  float*  negAll  = (float*)p;  p += (size_t)NB * 4;            // 64 KB
  __bf16* WtB     = (__bf16*)p; p += (size_t)NO * NDP * 2;      // 40 KB
  __bf16* WsTB    = (__bf16*)p; p += (size_t)NO * NO * 2;       // 8 KB
  __bf16* poolBp  = (__bf16*)p; p += (size_t)NB * NO * 2;       // 2 MB
  __bf16* poolBn  = (__bf16*)p; p += (size_t)NB * NO * 2;       // 2 MB
  __bf16* pAnchB  = (__bf16*)p; p += (size_t)NB * NO * 2;       // 2 MB
  __bf16* pPoolEB = (__bf16*)p; p += (size_t)NB * NO * 2;       // 2 MB
  __bf16* nPoolEB = (__bf16*)p; p += (size_t)NB * NO * 2;       // 2 MB

  // Stage 0: one-time bf16 repack of reused weight panels
  k_cvt_w<<<(NO * NDP + NO * NO + 255) / 256, 256, 0, stream>>>(Wm, Ws, WtB, WsTB);

  // Stage 1: feature GEMMs (TDM-staged W panel + LDS-staged A panel)
  k_gemm_feat<<<NN / 16, 128, 0, stream>>>(pos_feat, WtB, Xp);
  k_gemm_feat<<<NN / 16, 128, 0, stream>>>(neg_feat, WtB, Xn);

  // Stage 2: block-local 4x4 aggregation + pool (+ pos anchor embedding)
  k_block_reduce<<<NB / 4, 256, 0, stream>>>(Xp, pos_w, bias, prelu_a,
                                             poolBp, pAnchF, pAnchB);
  k_block_reduce<<<NB / 4, 256, 0, stream>>>(Xn, neg_w, bias, prelu_a,
                                             poolBn, (float*)nullptr, (__bf16*)nullptr);

  // Stage 3: pool embeddings (bf16 WMMA + row l2norm)
  k_pool_emb<<<NB / 16, 32, 0, stream>>>(poolBp, WsTB, bs, pPoolE, pPoolEB);
  k_pool_emb<<<NB / 16, 32, 0, stream>>>(poolBn, WsTB, bs, nPoolE, nPoolEB);

  // Stage 4: B x B similarity, exp-sum rows (dominant compute, bf16 WMMA)
  k_score<<<NB / 16, 256, 0, stream>>>(pAnchB, nPoolEB, negAll);

  // Stage 5: outputs
  k_final<<<NB / 256, 256, 0, stream>>>(pPoolE, pAnchF, negAll, out);
}
